// Encoder_90666759618598
// MI455X (gfx1250) — matrix operand order repaired and checked by reading, not compiled, not run on hardware
//
#include <hip/hip_runtime.h>

typedef __bf16  bf16;
typedef __attribute__((ext_vector_type(16))) __bf16 v16bf;
typedef __attribute__((ext_vector_type(8)))  __bf16 v8bf;
typedef __attribute__((ext_vector_type(8)))  float  v8f;

#define NN 50000
#define EE 800000
#define DD 128
#define ED 32
#define KF 288   // 2*DD + ED
#define LL 3
#define NEG_SLOPE 0.2f

// ---------------- small utility kernels ----------------

__global__ void zero_f32_kernel(float* p, int n) {
    int i = blockIdx.x * blockDim.x + threadIdx.x;
    if (i < n) p[i] = 0.0f;
}

__global__ void init_seg_kernel(unsigned* mx, float* ss, int n) {
    int i = blockIdx.x * blockDim.x + threadIdx.x;
    if (i < n) { mx[i] = 0u; ss[i] = 0.0f; }
}

__global__ void f32_to_bf16_kernel(const float* __restrict__ in, bf16* __restrict__ out, int n) {
    int i = blockIdx.x * blockDim.x + threadIdx.x;
    if (i < n) out[i] = (bf16)in[i];
}

// transpose fp32 [K][Nc] -> bf16 [Nc][K]
__global__ void transpose_bf_kernel(const float* __restrict__ in, bf16* __restrict__ out,
                                    int K, int Nc) {
    int o = blockIdx.x * blockDim.x + threadIdx.x;
    if (o >= K * Nc) return;
    int n = o / K, k = o % K;
    out[o] = (bf16)in[(size_t)k * Nc + n];
}

__global__ void copy_f32_kernel(const float* __restrict__ in, float* __restrict__ out, int n) {
    int i = blockIdx.x * blockDim.x + threadIdx.x;
    if (i < n) out[i] = in[i];
}

// ---------------- embeddings + batchnorm ----------------

__global__ void node_pre_kernel(const float* __restrict__ x, const float* __restrict__ tw,
                                const float* __restrict__ dem, const float* __restrict__ w,
                                const float* __restrict__ b, float* __restrict__ hpre,
                                float* __restrict__ sum, float* __restrict__ sq) {
    int d = threadIdx.x;                       // 128 threads
    int n0 = blockIdx.x * 64;
    float w0 = w[d], w1 = w[DD + d], w2 = w[2 * DD + d], w3 = w[3 * DD + d], w4 = w[4 * DD + d];
    float bb = b[d];
    float ls = 0.f, lq = 0.f;
    for (int i = 0; i < 64; ++i) {
        int n = n0 + i;
        if (n >= NN) break;
        float s = x[n * 2] * w0 + x[n * 2 + 1] * w1 + tw[n * 2] * w2 + tw[n * 2 + 1] * w3
                + dem[n] * w4 + bb;
        hpre[(size_t)n * DD + d] = s;
        ls += s; lq += s * s;
    }
    atomicAdd(&sum[d], ls);
    atomicAdd(&sq[d], lq);
}

__global__ void edge_pre_kernel(const float* __restrict__ ea, const float* __restrict__ w,
                                const float* __restrict__ b, float* __restrict__ epre,
                                float* __restrict__ sum, float* __restrict__ sq) {
    int tid = threadIdx.x;                     // 256 threads
    int d = tid & 31, sub = tid >> 5;          // 8 edge lanes
    float wd = w[d], bd = b[d];
    int e0 = blockIdx.x * 256 + sub;
    float ls = 0.f, lq = 0.f;
    for (int i = 0; i < 32; ++i) {
        int e = e0 + i * 8;
        if (e < EE) {
            float s = ea[e] * wd + bd;
            epre[(size_t)e * ED + d] = s;
            ls += s; lq += s * s;
        }
    }
    atomicAdd(&sum[d], ls);
    atomicAdd(&sq[d], lq);
}

__global__ void bn_finalize_kernel(const float* __restrict__ sum, const float* __restrict__ sq,
                                   const float* __restrict__ g, const float* __restrict__ bb,
                                   float* __restrict__ scale, float* __restrict__ shift,
                                   float invCount, int C) {
    int d = threadIdx.x;
    if (d >= C) return;
    float mean = sum[d] * invCount;
    float var  = sq[d] * invCount - mean * mean;
    float sc   = g[d] * rsqrtf(var + 1e-5f);
    scale[d] = sc;
    shift[d] = bb[d] - mean * sc;
}

__global__ void bn_apply_f32_kernel(const float* __restrict__ in, const float* __restrict__ scale,
                                    const float* __restrict__ shift, float* __restrict__ out,
                                    int n, int C) {
    int i = blockIdx.x * blockDim.x + threadIdx.x;
    if (i >= n) return;
    int d = i % C;
    out[i] = in[i] * scale[d] + shift[d];
}

__global__ void bn_apply_bf16_kernel(const float* __restrict__ in, const float* __restrict__ scale,
                                     const float* __restrict__ shift, bf16* __restrict__ out,
                                     int n, int C) {
    int i = blockIdx.x * blockDim.x + threadIdx.x;
    if (i >= n) return;
    int d = i % C;
    out[i] = (bf16)(in[i] * scale[d] + shift[d]);
}

// ---------------- WMMA GEMMs ----------------
// xt = h @ W + b  (N x 128 x 128), bf16 WMMA, fp32 accum
__global__ __launch_bounds__(256) void xt_gemm_kernel(
        const bf16* __restrict__ hb,       // [N][128] bf16
        const bf16* __restrict__ Bt,       // [128 cols][128 k] bf16 (W transposed)
        const float* __restrict__ bias,    // [128]
        float* __restrict__ xt, bf16* __restrict__ xtb, int nGroups) {
    int wave = threadIdx.x >> 5, lane = threadIdx.x & 31;
    int group = blockIdx.x * 8 + wave;
    if (group >= nGroups) return;
    int r = lane & 15, half = lane >> 4;
    int row = group * 16 + r;

    v8f acc[8];
    v8f z = {0.f, 0.f, 0.f, 0.f, 0.f, 0.f, 0.f, 0.f};
#pragma unroll
    for (int i = 0; i < 8; ++i) acc[i] = z;

    const bf16* arow = hb + (size_t)row * DD;
#pragma unroll
    for (int ks = 0; ks < 4; ++ks) {
        int k0 = ks * 32;
        v8bf c0 = *(const v8bf*)(arow + k0 + half * 16);
        v8bf c1 = *(const v8bf*)(arow + k0 + half * 16 + 8);
        v16bf a;
#pragma unroll
        for (int i = 0; i < 8; ++i) { a[i] = c0[i]; a[i + 8] = c1[i]; }
#pragma unroll
        for (int nt = 0; nt < 8; ++nt) {
            int col = nt * 16 + r;
            v16bf bfr = *(const v16bf*)(Bt + (size_t)col * DD + k0 + half * 16);
            acc[nt] = __builtin_amdgcn_wmma_f32_16x16x32_bf16(
                false, a, false, bfr, (short)0, acc[nt], false, false);
        }
    }
#pragma unroll
    for (int rr = 0; rr < 8; ++rr) {
        int m = rr + half * 8;
        size_t orow = (size_t)(group * 16 + m) * DD;
#pragma unroll
        for (int nt = 0; nt < 8; ++nt) {
            int col = nt * 16 + r;
            float v = acc[nt][rr] + bias[col];
            xt[orow + col]  = v;
            xtb[orow + col] = (bf16)v;
        }
    }
}

// alpha = leakyrelu([xt[dst] | xt[src] | e] @ attn_w + b); also segment-max via u32 atomics
__device__ __forceinline__ v8bf load_feat(const bf16* pi, const bf16* pj, const bf16* pe, int kk) {
    const bf16* p;
    if (kk < DD)            p = pi + kk;
    else if (kk < 2 * DD)   p = pj + (kk - DD);
    else                    p = pe + (kk - 2 * DD);
    return *(const v8bf*)p;
}

__global__ __launch_bounds__(256) void attn_gemm_kernel(
        const bf16* __restrict__ xtb,      // [N][128] bf16
        const bf16* __restrict__ eb,       // [E][32] bf16
        const bf16* __restrict__ Bt,       // [128 cols][288 k] bf16 (attn_w transposed)
        const float* __restrict__ bias,    // [128]
        const int* __restrict__ src, const int* __restrict__ dst,
        float* __restrict__ alpha, unsigned* __restrict__ mxEnc, int nGroups) {
    int wave = threadIdx.x >> 5, lane = threadIdx.x & 31;
    int group = blockIdx.x * 8 + wave;
    if (group >= nGroups) return;
    int r = lane & 15, half = lane >> 4;
    int row = group * 16 + r;
    int js = src[row], jd = dst[row];
    const bf16* pi = xtb + (size_t)jd * DD;
    const bf16* pj = xtb + (size_t)js * DD;
    const bf16* pe = eb + (size_t)row * ED;

    v8f acc[8];
    v8f z = {0.f, 0.f, 0.f, 0.f, 0.f, 0.f, 0.f, 0.f};
#pragma unroll
    for (int i = 0; i < 8; ++i) acc[i] = z;

#pragma unroll
    for (int ks = 0; ks < 9; ++ks) {
        int k0 = ks * 32;
        v8bf c0 = load_feat(pi, pj, pe, k0 + half * 16);
        v8bf c1 = load_feat(pi, pj, pe, k0 + half * 16 + 8);
        v16bf a;
#pragma unroll
        for (int i = 0; i < 8; ++i) { a[i] = c0[i]; a[i + 8] = c1[i]; }
#pragma unroll
        for (int nt = 0; nt < 8; ++nt) {
            int col = nt * 16 + r;
            v16bf bfr = *(const v16bf*)(Bt + (size_t)col * KF + k0 + half * 16);
            acc[nt] = __builtin_amdgcn_wmma_f32_16x16x32_bf16(
                false, a, false, bfr, (short)0, acc[nt], false, false);
        }
    }
#pragma unroll
    for (int rr = 0; rr < 8; ++rr) {
        int m = rr + half * 8;
        int grow = group * 16 + m;
        int dm = dst[grow];
        size_t arow = (size_t)grow * DD;
        size_t mrow = (size_t)dm * DD;
#pragma unroll
        for (int nt = 0; nt < 8; ++nt) {
            int col = nt * 16 + r;
            float v = acc[nt][rr] + bias[col];
            v = v > 0.f ? v : NEG_SLOPE * v;
            alpha[arow + col] = v;
            unsigned bits = __float_as_uint(v);
            unsigned enc = (bits & 0x80000000u) ? ~bits : (bits | 0x80000000u);
            atomicMax(&mxEnc[mrow + col], enc);
        }
    }
}

// ---------------- segment softmax + aggregation ----------------

__global__ void exp_ssum_kernel(float* __restrict__ alpha, const unsigned* __restrict__ mxEnc,
                                const int* __restrict__ dst, float* __restrict__ ssum) {
    int e = blockIdx.x, d = threadIdx.x;     // 128 threads
    int jd = dst[e];
    unsigned enc = mxEnc[(size_t)jd * DD + d];
    unsigned mb = (enc & 0x80000000u) ? (enc ^ 0x80000000u) : ~enc;
    float m = __uint_as_float(mb);
    float v = __expf(alpha[(size_t)e * DD + d] - m);
    alpha[(size_t)e * DD + d] = v;
    atomicAdd(&ssum[(size_t)jd * DD + d], v);
}

__global__ void aggregate_kernel(const float* __restrict__ alpha, const float* __restrict__ ssum,
                                 const float* __restrict__ xt, const int* __restrict__ src,
                                 const int* __restrict__ dst, float* __restrict__ h) {
    int e = blockIdx.x, d = threadIdx.x;     // 128 threads
    int jd = dst[e], js = src[e];
    float a = alpha[(size_t)e * DD + d] / (ssum[(size_t)jd * DD + d] + 1e-16f);
    atomicAdd(&h[(size_t)jd * DD + d], xt[(size_t)js * DD + d] * a);
}

__global__ void col_mean_kernel(const float* __restrict__ h, float* __restrict__ outp) {
    __shared__ float sm[256];
    int d = blockIdx.x, t = threadIdx.x;
    float s = 0.f;
    for (int n = t; n < NN; n += 256) s += h[(size_t)n * DD + d];
    sm[t] = s;
    __syncthreads();
    for (int o = 128; o > 0; o >>= 1) {
        if (t < o) sm[t] += sm[t + o];
        __syncthreads();
    }
    if (t == 0) outp[d] = sm[0] / (float)NN;
}

// ---------------- host driver ----------------

extern "C" void kernel_launch(void* const* d_in, const int* in_sizes, int n_in,
                              void* d_out, int out_size, void* d_ws, size_t ws_size,
                              hipStream_t stream) {
    const float* x         = (const float*)d_in[0];
    const float* tw        = (const float*)d_in[1];
    const float* demand    = (const float*)d_in[2];
    const float* edge_attr = (const float*)d_in[3];
    const int*   eidx      = (const int*)d_in[4];
    const int*   src       = eidx;
    const int*   dst       = eidx + EE;
    const float* fc_node_w = (const float*)d_in[5];
    const float* fc_node_b = (const float*)d_in[6];
    const float* bn_g      = (const float*)d_in[7];
    const float* bn_b      = (const float*)d_in[8];
    const float* fc_edge_w = (const float*)d_in[9];
    const float* fc_edge_b = (const float*)d_in[10];
    const float* be_g      = (const float*)d_in[11];
    const float* be_b      = (const float*)d_in[12];
    const float* conv_fc_w   = (const float*)d_in[13];
    const float* conv_fc_b   = (const float*)d_in[14];
    const float* conv_attn_w = (const float*)d_in[15];
    const float* conv_attn_b = (const float*)d_in[16];
    float* out = (float*)d_out;

    char* ws = (char*)d_ws;
    size_t off = 0;
    auto alloc = [&](size_t bytes) -> char* {
        char* p = ws + off;
        off += (bytes + 255) & ~(size_t)255;
        return p;
    };
    float*    h     = (float*)alloc((size_t)NN * DD * 4);
    bf16*     hb    = (bf16*)alloc((size_t)NN * DD * 2);
    float*    xt    = (float*)alloc((size_t)NN * DD * 4);   // also h_pre before layers
    bf16*     xtb   = (bf16*)alloc((size_t)NN * DD * 2);
    bf16*     ebf   = (bf16*)alloc((size_t)EE * ED * 2);
    unsigned* mxEnc = (unsigned*)alloc((size_t)NN * DD * 4);
    float*    ssum  = (float*)alloc((size_t)NN * DD * 4);
    float*    alpha = (float*)alloc((size_t)EE * DD * 4);   // e_pre reuses the front of this
    bf16*     BtAttn = (bf16*)alloc((size_t)DD * KF * 2);
    bf16*     BtFc   = (bf16*)alloc((size_t)DD * DD * 2);
    float*    stats  = (float*)alloc(1024 * 4);
    float* sumN = stats,        * sqN = stats + 128;
    float* sumE = stats + 256,  * sqE = stats + 288;
    float* scN  = stats + 320,  * shN = stats + 448;
    float* scE  = stats + 576,  * shE = stats + 608;
    float* e_pre = alpha;

    const int ND = NN * DD;            // 6.4M
    const int EDtot = EE * ED;         // 25.6M
    const int EDD = EE * DD;           // 102.4M (fits int)
    const int nGroupsN = NN / 16;      // 3125
    const int nGroupsE = EE / 16;      // 50000

    // --- embeddings + batchnorm ---
    zero_f32_kernel<<<2, 256, 0, stream>>>(stats, 320);
    node_pre_kernel<<<(NN + 63) / 64, 128, 0, stream>>>(x, tw, demand, fc_node_w, fc_node_b,
                                                        xt /*h_pre*/, sumN, sqN);
    bn_finalize_kernel<<<1, 128, 0, stream>>>(sumN, sqN, bn_g, bn_b, scN, shN, 1.0f / NN, 128);
    bn_apply_f32_kernel<<<(ND + 255) / 256, 256, 0, stream>>>(xt, scN, shN, h, ND, DD);

    edge_pre_kernel<<<EE / 256, 256, 0, stream>>>(edge_attr, fc_edge_w, fc_edge_b, e_pre, sumE, sqE);
    bn_finalize_kernel<<<1, 32, 0, stream>>>(sumE, sqE, be_g, be_b, scE, shE, 1.0f / EE, 32);
    bn_apply_bf16_kernel<<<(EDtot + 255) / 256, 256, 0, stream>>>(e_pre, scE, shE, ebf, EDtot, ED);

    // --- GAT layers ---
    for (int l = 0; l < LL; ++l) {
        transpose_bf_kernel<<<(DD * DD + 255) / 256, 256, 0, stream>>>(
            conv_fc_w + (size_t)l * DD * DD, BtFc, DD, DD);
        transpose_bf_kernel<<<(KF * DD + 255) / 256, 256, 0, stream>>>(
            conv_attn_w + (size_t)l * KF * DD, BtAttn, KF, DD);
        f32_to_bf16_kernel<<<(ND + 255) / 256, 256, 0, stream>>>(h, hb, ND);

        xt_gemm_kernel<<<(nGroupsN + 7) / 8, 256, 0, stream>>>(
            hb, BtFc, conv_fc_b + (size_t)l * DD, xt, xtb, nGroupsN);

        init_seg_kernel<<<(ND + 255) / 256, 256, 0, stream>>>(mxEnc, ssum, ND);

        attn_gemm_kernel<<<(nGroupsE + 7) / 8, 256, 0, stream>>>(
            xtb, ebf, BtAttn, conv_attn_b + (size_t)l * DD, src, dst, alpha, mxEnc, nGroupsE);

        exp_ssum_kernel<<<EE, 128, 0, stream>>>(alpha, mxEnc, dst, ssum);
        aggregate_kernel<<<EE, 128, 0, stream>>>(alpha, ssum, xt, src, dst, h);
    }

    // --- outputs: [N*D] h, then [D] column mean ---
    copy_f32_kernel<<<(ND + 255) / 256, 256, 0, stream>>>(h, out, ND);
    col_mean_kernel<<<128, 256, 0, stream>>>(h, out + (size_t)NN * DD);
    (void)in_sizes; (void)n_in; (void)out_size; (void)ws_size; (void)EDD;
}
